// Glm4MoeForCausalLM_85255100825932
// MI455X (gfx1250) — matrix operand order repaired and checked by reading, not compiled, not run on hardware
//
#include <hip/hip_runtime.h>
#include <hip/hip_bf16.h>
#include <math.h>

#define NUM_EXPERTS 8
#define D_MODEL 2048
#define D_FFN   1408
#define T_TOKENS 2048
#define NSLOT 9            // 8 routed experts + 1 shared
#define BM 64
#define BN 64
#define BK 32
#define LDS_STRIDE 40      // bf16 elems per LDS row (80B = 64B data + 16B pad)

typedef __bf16 bf16_t;
typedef __attribute__((ext_vector_type(16))) __bf16 v16bf;
typedef __attribute__((ext_vector_type(8)))  __bf16 v8bf;
typedef __attribute__((ext_vector_type(8)))  float  v8f;
typedef __attribute__((ext_vector_type(4)))  float  v4f;
typedef __attribute__((ext_vector_type(4)))  int    v4i;

#if defined(__HIP_DEVICE_COMPILE__) && __has_builtin(__builtin_amdgcn_global_load_async_to_lds_b128)
#define USE_ASYNC 1
// pointer-to-addrspace(1)-v4i and pointer-to-addrspace(3)-v4i
typedef __attribute__((address_space(1))) v4i* gv4i_p;
typedef __attribute__((address_space(3))) v4i* lv4i_p;
#else
#define USE_ASYNC 0
#endif

// ---------------------------------------------------------------------------
// Kernel 1: router — logits -> softmax -> top-2 -> renormalized dense gates
// ---------------------------------------------------------------------------
__global__ __launch_bounds__(256) void moe_router(const float* __restrict__ x,
                                                  const float* __restrict__ gw,
                                                  float* __restrict__ gates) {
  int wave = threadIdx.x >> 5;
  int lane = threadIdx.x & 31;
  int t = blockIdx.x * 8 + wave;
  if (t >= T_TOKENS) return;

  float acc[NUM_EXPERTS];
#pragma unroll
  for (int e = 0; e < NUM_EXPERTS; ++e) acc[e] = 0.f;

  const float* xr = x + (size_t)t * D_MODEL;
  for (int d = lane; d < D_MODEL; d += 32) {
    float xv = xr[d];
#pragma unroll
    for (int e = 0; e < NUM_EXPERTS; ++e) acc[e] += xv * gw[e * D_MODEL + d];
  }
#pragma unroll
  for (int e = 0; e < NUM_EXPERTS; ++e) {
#pragma unroll
    for (int off = 16; off > 0; off >>= 1) acc[e] += __shfl_xor(acc[e], off, 32);
  }
  if (lane == 0) {
    float mx = acc[0];
#pragma unroll
    for (int e = 1; e < NUM_EXPERTS; ++e) mx = fmaxf(mx, acc[e]);
    float p[NUM_EXPERTS];
#pragma unroll
    for (int e = 0; e < NUM_EXPERTS; ++e) p[e] = __expf(acc[e] - mx);
    // top-2 (softmax normalization cancels in the top-k renormalization)
    int i0 = 0;
#pragma unroll
    for (int e = 1; e < NUM_EXPERTS; ++e) if (p[e] > p[i0]) i0 = e;
    int i1 = (i0 == 0) ? 1 : 0;
#pragma unroll
    for (int e = 0; e < NUM_EXPERTS; ++e) if (e != i0 && p[e] > p[i1]) i1 = e;
    float rs = 1.f / (p[i0] + p[i1]);
#pragma unroll
    for (int e = 0; e < NUM_EXPERTS; ++e) gates[(size_t)t * NUM_EXPERTS + e] = 0.f;
    gates[(size_t)t * NUM_EXPERTS + i0] = p[i0] * rs;
    gates[(size_t)t * NUM_EXPERTS + i1] = p[i1] * rs;
  }
}

// ---------------------------------------------------------------------------
// WMMA fragment loaders (per documented CDNA5 bf16 VGPR layouts)
// ---------------------------------------------------------------------------
__device__ inline v16bf load_afrag(const bf16_t* lA, int row, int half) {
  const bf16_t* base = lA + row * LDS_STRIDE;
  v8bf lo = *(const v8bf*)(base + half * 8);        // K = half*8 .. +7
  v8bf hi = *(const v8bf*)(base + 16 + half * 8);   // K = 16+half*8 .. +7
  v16bf a;
#pragma unroll
  for (int i = 0; i < 8; ++i) { a[i] = lo[i]; a[8 + i] = hi[i]; }
  return a;
}
__device__ inline v16bf load_bfrag(const bf16_t* lB, int n, int half) {
  const bf16_t* base = lB + n * LDS_STRIDE;  // K = half*8 .. +7, 16+half*8 .. +7
  v8bf lo = *(const v8bf*)(base + half * 8);
  v8bf hi = *(const v8bf*)(base + 16 + half * 8);
  v16bf b;
#pragma unroll
  for (int i = 0; i < 8; ++i) { b[i] = lo[i]; b[8 + i] = hi[i]; }
  return b;
}

// Stage [BM x BK] f32 activation tile -> LDS bf16 (row-major, padded rows)
__device__ inline void stage_x_tile(bf16_t* lA, const float* X, int ldx,
                                    int m0, int k0, int tid) {
  int r = tid >> 1, half = tid & 1;
  const float* src = X + (size_t)(m0 + r) * ldx + k0 + half * 16;
  v4f f0 = *(const v4f*)(src);
  v4f f1 = *(const v4f*)(src + 4);
  v4f f2 = *(const v4f*)(src + 8);
  v4f f3 = *(const v4f*)(src + 12);
  v8bf a, b;
#pragma unroll
  for (int i = 0; i < 4; ++i) {
    a[i] = (__bf16)f0[i]; a[4 + i] = (__bf16)f1[i];
    b[i] = (__bf16)f2[i]; b[4 + i] = (__bf16)f3[i];
  }
  *(v8bf*)(lA + r * LDS_STRIDE + half * 16) = a;
  *(v8bf*)(lA + r * LDS_STRIDE + half * 16 + 8) = b;
}

// Stage [BK x BN] f32 weight tile (row-major, row stride ldw) transposed into
// LDS as [BN][BK] bf16. Global reads coalesced across n; LDS writes 16B.
__device__ inline void stage_w_tile(bf16_t* lB, const float* W, int ldw,
                                    int k0, int n0, int item /*0..255*/) {
  int n = item & 63;
  int kg = (item >> 6) & 3;
  const float* src = W + (size_t)(k0 + kg * 8) * ldw + n0 + n;
  v8bf v;
#pragma unroll
  for (int i = 0; i < 8; ++i) v[i] = (__bf16)src[(size_t)i * ldw];
  *(v8bf*)(lB + n * LDS_STRIDE + kg * 8) = v;
}

// Stage [BM x BK] bf16 H tile -> LDS (raw copy; async path if available)
__device__ inline void stage_h_tile(bf16_t* lA, const bf16_t* Ht,
                                    int m0, int k0, int tid) {
#if USE_ASYNC
#pragma unroll
  for (int it = 0; it < 2; ++it) {
    int c = it * 128 + tid;        // 256 x 16B chunks
    int r = c >> 2, q = c & 3;
    const bf16_t* src = Ht + (size_t)(m0 + r) * D_FFN + k0 + q * 8;
    bf16_t* dst = lA + r * LDS_STRIDE + q * 8;
    __builtin_amdgcn_global_load_async_to_lds_b128((gv4i_p)src, (lv4i_p)dst, 0, 0);
  }
#else
  int r = tid >> 1, half = tid & 1;
  const bf16_t* src = Ht + (size_t)(m0 + r) * D_FFN + k0 + half * 16;
  v8bf v0 = *(const v8bf*)src;
  v8bf v1 = *(const v8bf*)(src + 8);
  *(v8bf*)(lA + r * LDS_STRIDE + half * 16) = v0;
  *(v8bf*)(lA + r * LDS_STRIDE + half * 16 + 8) = v1;
#endif
}

__device__ inline void async_wait() {
#if USE_ASYNC && __has_builtin(__builtin_amdgcn_s_wait_asynccnt)
  __builtin_amdgcn_s_wait_asynccnt(0);
#endif
}

#define WMMA_BF16(A, B, C) \
  __builtin_amdgcn_wmma_f32_16x16x32_bf16(false, (A), false, (B), (short)0, (C), false, false)

// ---------------------------------------------------------------------------
// Kernel 2: H[es] = gate_scale * silu(x@Wg) * (x@Wu)   (bf16, gate-pre-scaled)
// ---------------------------------------------------------------------------
__global__ __launch_bounds__(128) void moe_gateup(
    const float* __restrict__ x,
    const float* __restrict__ w_gate, const float* __restrict__ w_up,
    const float* __restrict__ sw_gate, const float* __restrict__ sw_up,
    const float* __restrict__ gates, bf16_t* __restrict__ H) {
  __shared__ __align__(16) bf16_t lA[BM * LDS_STRIDE];
  __shared__ __align__(16) bf16_t lBg[BN * LDS_STRIDE];
  __shared__ __align__(16) bf16_t lBu[BN * LDS_STRIDE];
  __shared__ int s_active;

  const int es = blockIdx.z;
  const int tM = blockIdx.y * BM;
  const int tN = blockIdx.x * BN;
  const int tid = threadIdx.x;

  if (tid == 0) s_active = (es == 8) ? 1 : 0;
  __syncthreads();
  if (es < 8 && tid < BM && gates[(size_t)(tM + tid) * NUM_EXPERTS + es] != 0.f)
    s_active = 1;
  __syncthreads();
  if (!s_active) return;   // uniform: whole block exits together

  const float* Wg = (es < 8) ? w_gate + (size_t)es * D_MODEL * D_FFN : sw_gate;
  const float* Wu = (es < 8) ? w_up   + (size_t)es * D_MODEL * D_FFN : sw_up;

  const int wave = tid >> 5, lane = tid & 31;
  const int wm = (wave >> 1) * 32, wn = (wave & 1) * 32;
  const int half = lane >> 4, lr = lane & 15;

  v8f accg[2][2] = {};
  v8f accu[2][2] = {};

  for (int k0 = 0; k0 < D_MODEL; k0 += BK) {
    stage_x_tile(lA, x, D_MODEL, tM, k0, tid);
#pragma unroll
    for (int it = 0; it < 4; ++it) {
      int item = it * 128 + tid;                       // 0..511
      stage_w_tile((item & 256) ? lBu : lBg,
                   (item & 256) ? Wu : Wg, D_FFN, k0, tN, item & 255);
    }
    __syncthreads();

    v16bf a0 = load_afrag(lA, wm + lr, half);
    v16bf a1 = load_afrag(lA, wm + 16 + lr, half);
    v16bf g0 = load_bfrag(lBg, wn + lr, half);
    v16bf g1 = load_bfrag(lBg, wn + 16 + lr, half);
    v16bf u0 = load_bfrag(lBu, wn + lr, half);
    v16bf u1 = load_bfrag(lBu, wn + 16 + lr, half);

    accg[0][0] = WMMA_BF16(a0, g0, accg[0][0]);
    accg[0][1] = WMMA_BF16(a0, g1, accg[0][1]);
    accg[1][0] = WMMA_BF16(a1, g0, accg[1][0]);
    accg[1][1] = WMMA_BF16(a1, g1, accg[1][1]);
    accu[0][0] = WMMA_BF16(a0, u0, accu[0][0]);
    accu[0][1] = WMMA_BF16(a0, u1, accu[0][1]);
    accu[1][0] = WMMA_BF16(a1, u0, accu[1][0]);
    accu[1][1] = WMMA_BF16(a1, u1, accu[1][1]);
    __syncthreads();
  }

  bf16_t* Ht = H + (size_t)es * T_TOKENS * D_FFN;
#pragma unroll
  for (int mi = 0; mi < 2; ++mi) {
    float gsc[8];
#pragma unroll
    for (int i = 0; i < 8; ++i) {
      int t = tM + wm + mi * 16 + half * 8 + i;
      gsc[i] = (es < 8) ? gates[(size_t)t * NUM_EXPERTS + es] : 1.0f;
    }
#pragma unroll
    for (int ni = 0; ni < 2; ++ni) {
      v8f g = accg[mi][ni];
      v8f u = accu[mi][ni];
#pragma unroll
      for (int i = 0; i < 8; ++i) {
        int t = tM + wm + mi * 16 + half * 8 + i;
        int f = tN + wn + ni * 16 + lr;
        float gv = g[i];
        float hv = (gv / (1.f + __expf(-gv))) * u[i] * gsc[i];
        Ht[(size_t)t * D_FFN + f] = (__bf16)hv;
      }
    }
  }
}

// ---------------------------------------------------------------------------
// Kernel 3: out[t,d] = sum_es H[es][t,:] @ Wd[es][:,d]  (H gate-pre-scaled)
// ---------------------------------------------------------------------------
__global__ __launch_bounds__(128) void moe_down(
    const bf16_t* __restrict__ H,
    const float* __restrict__ w_down, const float* __restrict__ sw_down,
    const float* __restrict__ gates, float* __restrict__ out) {
  __shared__ __align__(16) bf16_t lA[BM * LDS_STRIDE];
  __shared__ __align__(16) bf16_t lB[BN * LDS_STRIDE];
  __shared__ int s_active;

  const int tM = blockIdx.y * BM;
  const int tN = blockIdx.x * BN;
  const int tid = threadIdx.x;
  const int wave = tid >> 5, lane = tid & 31;
  const int wm = (wave >> 1) * 32, wn = (wave & 1) * 32;
  const int half = lane >> 4, lr = lane & 15;

  v8f acc[2][2] = {};

  for (int es = 0; es < NSLOT; ++es) {
    if (tid == 0) s_active = (es == 8) ? 1 : 0;
    __syncthreads();
    if (es < 8 && tid < BM && gates[(size_t)(tM + tid) * NUM_EXPERTS + es] != 0.f)
      s_active = 1;
    __syncthreads();
    if (!s_active) continue;   // uniform per block

    const bf16_t* Ht = H + (size_t)es * T_TOKENS * D_FFN;
    const float* Wd = (es < 8) ? w_down + (size_t)es * D_FFN * D_MODEL : sw_down;

    for (int k0 = 0; k0 < D_FFN; k0 += BK) {
      stage_h_tile(lA, Ht, tM, k0, tid);
#pragma unroll
      for (int it = 0; it < 2; ++it)
        stage_w_tile(lB, Wd, D_MODEL, k0, tN, it * 128 + tid);
      async_wait();
      __syncthreads();

      v16bf a0 = load_afrag(lA, wm + lr, half);
      v16bf a1 = load_afrag(lA, wm + 16 + lr, half);
      v16bf b0 = load_bfrag(lB, wn + lr, half);
      v16bf b1 = load_bfrag(lB, wn + 16 + lr, half);

      acc[0][0] = WMMA_BF16(a0, b0, acc[0][0]);
      acc[0][1] = WMMA_BF16(a0, b1, acc[0][1]);
      acc[1][0] = WMMA_BF16(a1, b0, acc[1][0]);
      acc[1][1] = WMMA_BF16(a1, b1, acc[1][1]);
      __syncthreads();
    }
  }

#pragma unroll
  for (int mi = 0; mi < 2; ++mi)
#pragma unroll
    for (int ni = 0; ni < 2; ++ni)
#pragma unroll
      for (int i = 0; i < 8; ++i) {
        int t = tM + wm + mi * 16 + half * 8 + i;
        int n = tN + wn + ni * 16 + lr;
        out[(size_t)t * D_MODEL + n] = acc[mi][ni][i];
      }
}

// ---------------------------------------------------------------------------
extern "C" void kernel_launch(void* const* d_in, const int* in_sizes, int n_in,
                              void* d_out, int out_size, void* d_ws, size_t ws_size,
                              hipStream_t stream) {
  (void)in_sizes; (void)n_in; (void)out_size; (void)ws_size;
  const float* x       = (const float*)d_in[0];
  const float* gate_w  = (const float*)d_in[1];
  const float* w_gate  = (const float*)d_in[2];
  const float* w_up    = (const float*)d_in[3];
  const float* w_down  = (const float*)d_in[4];
  const float* sw_gate = (const float*)d_in[5];
  const float* sw_up   = (const float*)d_in[6];
  const float* sw_down = (const float*)d_in[7];
  float* out = (float*)d_out;

  // workspace: [gates: T x 8 f32][H: 9 x T x F bf16]
  float* gates = (float*)d_ws;
  bf16_t* H = (bf16_t*)((char*)d_ws + (size_t)T_TOKENS * NUM_EXPERTS * sizeof(float));

  moe_router<<<T_TOKENS / 8, 256, 0, stream>>>(x, gate_w, gates);

  dim3 g2(D_FFN / BN, T_TOKENS / BM, NSLOT);
  moe_gateup<<<g2, 128, 0, stream>>>(x, w_gate, w_up, sw_gate, sw_up, gates, H);

  dim3 g3(D_MODEL / BN, T_TOKENS / BM);
  moe_down<<<g3, 128, 0, stream>>>(H, w_down, sw_down, gates, out);
}
